// DetModel_78529182040708
// MI455X (gfx1250) — matrix operand order repaired and checked by reading, not compiled, not run on hardware
//
#include <hip/hip_runtime.h>
#include <hip/hip_bf16.h>
#include <stdint.h>
#include <stddef.h>

typedef _Float16 v16h __attribute__((ext_vector_type(16)));
typedef _Float16 v8h  __attribute__((ext_vector_type(8)));
typedef float    v8f  __attribute__((ext_vector_type(8)));
typedef int      v4i  __attribute__((ext_vector_type(4)));

// ---- gfx1250 async (memory -> LDS) path, guarded so absence degrades safely ----
#if __has_builtin(__builtin_amdgcn_global_load_async_to_lds_b128)
#define ATHENA_HAVE_ASYNC 1
#else
#define ATHENA_HAVE_ASYNC 0
#endif

__device__ __forceinline__ void async_cp16(const void* g, void* l) {
#if ATHENA_HAVE_ASYNC && defined(__HIP_DEVICE_COMPILE__)
    // emits global_load_async_to_lds_b128 (ASYNCcnt-tracked, bypasses VGPRs)
    // signature (from compiler diagnostic): (int4 as(1)*, int4 as(3)*, imm, imm)
    __builtin_amdgcn_global_load_async_to_lds_b128(
        (__attribute__((address_space(1))) v4i*)(g),
        (__attribute__((address_space(3))) v4i*)(l), 0, 0);
#else
    *(v8h*)l = *(const v8h*)g;
#endif
}

__device__ __forceinline__ void wait_async() {
#if ATHENA_HAVE_ASYNC && defined(__HIP_DEVICE_COMPILE__)
#if __has_builtin(__builtin_amdgcn_s_wait_asynccnt)
    __builtin_amdgcn_s_wait_asynccnt(0);
#else
    asm volatile("s_wait_asynccnt 0x0" ::: "memory");
#endif
#endif
}

static constexpr int BATCH = 4, HH = 512, WW = 512;
static constexpr long long NPIX = (long long)BATCH * HH * WW;   // 1,048,576

// ---------------- workspace layout (bytes) ----------------
static constexpr size_t WS_BUF1    = 0;                                   // f16 [B,H,W,64]
static constexpr size_t WS_BUF2    = WS_BUF1 + (size_t)NPIX * 64 * 2;     // f16 [B,H,W,64]
static constexpr size_t WS_OFFF    = WS_BUF2 + (size_t)NPIX * 64 * 2;     // f32 [B,H,W,6]
static constexpr size_t WS_WFULL   = WS_OFFF + (size_t)NPIX * 6 * 4;      // f32 [B,H,W,3]
static constexpr size_t SZ_P1 = (size_t)9 * 32 * 64 * 2;   // packed conv1 weights (f16)
static constexpr size_t SZ_P2 = (size_t)9 * 64 * 64 * 2;
static constexpr size_t SZ_P3 = (size_t)9 * 64 * 16 * 2;   // conv3 padded to 16 out ch
static constexpr size_t WS_PACK_O1 = WS_WFULL + (size_t)NPIX * 3 * 4;
static constexpr size_t WS_PACK_O2 = WS_PACK_O1 + SZ_P1;
static constexpr size_t WS_PACK_O3 = WS_PACK_O2 + SZ_P2;
static constexpr size_t WS_PACK_W1 = WS_PACK_O3 + SZ_P3;
static constexpr size_t WS_PACK_W2 = WS_PACK_W1 + SZ_P1;
static constexpr size_t WS_PACK_W3 = WS_PACK_W2 + SZ_P2;
static constexpr size_t WS_CLS     = WS_PACK_W3 + SZ_P3;

// ---------------------------------------------------------------------------
// Pack HWIO f32 weights into per-wave B-fragment order for v_wmma_f32_16x16x32_f16.
// Layout: [tap 0..8][kchunk][ntile][lane 0..31][half 0..15]  (1 KB per fragment)
//   lane < 16 : K = kc*32 + j,       N = lane
//   lane >= 16: K = kc*32 + 16 + j,  N = lane-16
// so each lane reads its 16 halves contiguously (2 x b128) at WMMA time.
// ---------------------------------------------------------------------------
__global__ __launch_bounds__(256) void pack_w_kernel(const float* __restrict__ W,
                                                     _Float16* __restrict__ dst,
                                                     int CIN, int COUT_REAL, int NTILES) {
    int total = 9 * (CIN / 32) * NTILES * 512;
    int i = blockIdx.x * blockDim.x + threadIdx.x;
    if (i >= total) return;
    int j    = i & 15;
    int lane = (i >> 4) & 31;
    int rest = i >> 9;
    int ntile = rest % NTILES;
    int kc    = (rest / NTILES) % (CIN / 32);
    int tap   = rest / (NTILES * (CIN / 32));
    int K = kc * 32 + ((lane >= 16) ? 16 : 0) + j;
    int n = ntile * 16 + (lane & 15);
    float v = (n < COUT_REAL) ? W[((size_t)tap * CIN + K) * COUT_REAL + n] : 0.0f;
    dst[i] = (_Float16)v;
}

// ---------------------------------------------------------------------------
// Implicit-GEMM 3x3 SAME conv, wave32 WMMA f16->f32.
//   CIN      : input channels (32 or 64); f16, or f32 converted during staging
//   PIXTILES : 16-pixel tiles along W per workgroup (PIXTILES*NTILES must == 8)
//   NTILES   : 16-out-channel tiles (COUT = 16*NTILES)
//   FINAL    : write f32 (stride COUT_REAL), no ReLU;  else write f16, ReLU
// f16 inputs are staged to LDS with global_load_async_to_lds_b128 when available.
// ---------------------------------------------------------------------------
template <int CIN, int PIXTILES, int NTILES, int COUT_REAL, bool FINAL, bool IN_F32>
__global__ __launch_bounds__(256) void conv3x3_kernel(const void* __restrict__ in,
                                                      const _Float16* __restrict__ wpack,
                                                      const float* __restrict__ bias,
                                                      _Float16* __restrict__ out16,
                                                      float* __restrict__ outf) {
    constexpr int TPIX = PIXTILES * 16;
    constexpr int PIXW = TPIX + 2;          // +halo
    constexpr int COUT = NTILES * 16;

    __shared__ _Float16 tile[3][PIXW][CIN];

    // decode block -> (batch, row, x-tile)
    int tilesPerRow = WW / TPIX;
    int bid = blockIdx.x;
    int tx  = bid % tilesPerRow;
    int rem = bid / tilesPerRow;
    int y   = rem % HH;
    int b   = rem / HH;
    int xbase = tx * TPIX;

    // hint: pull packed weights toward the caches (gfx1250 global_prefetch_b8)
    __builtin_prefetch(wpack + (threadIdx.x & 31) * 16, 0, 1);

    // ---- stage input tile (3 rows x PIXW pixels x CIN ch) into LDS ----
    constexpr int CPR = CIN / 8;            // 8-half chunks per pixel
    constexpr int CHUNKS = 3 * PIXW * CPR;
    for (int i = threadIdx.x; i < CHUNKS; i += 256) {
        int c8 = i % CPR;
        int p  = (i / CPR) % PIXW;
        int r  = i / (CPR * PIXW);
        int yy = y + r - 1;
        int xx = xbase + p - 1;
        bool inb = ((unsigned)yy < (unsigned)HH) && ((unsigned)xx < (unsigned)WW);
        _Float16* dst = &tile[r][p][c8 * 8];
        size_t base = (((size_t)b * HH + (inb ? yy : 0)) * WW + (inb ? xx : 0)) * CIN + c8 * 8;
        if constexpr (!IN_F32) {
            const _Float16* gin = (const _Float16*)in;
            if (ATHENA_HAVE_ASYNC) {
                if (inb) {
                    async_cp16(gin + base, dst);        // memory -> LDS, no VGPRs
                } else {
                    v8h z = {};
                    *(v8h*)dst = z;                      // halo: zero-fill via ds_store
                }
            } else {
                v8h val = {};
                if (inb) val = *(const v8h*)(gin + base);
                *(v8h*)dst = val;
            }
        } else {
            v8h val = {};
            if (inb) {
                const float* gin = (const float*)in;
                v8f t = *(const v8f*)(gin + base);
#pragma unroll
                for (int j = 0; j < 8; ++j) val[j] = (_Float16)t[j];
            }
            *(v8h*)dst = val;
        }
    }
    wait_async();          // s_wait_asynccnt 0: this wave's async LDS writes landed
    __syncthreads();

    // ---- per-wave 16-pixel x 16-outchannel WMMA tile ----
    int wave = threadIdx.x >> 5;
    int lane = threadIdx.x & 31;
    int pixT  = wave % PIXTILES;
    int ntile = wave / PIXTILES;
    int m  = lane & 15;
    int hi = lane >> 4;                     // 0: K {0-15}  1: K {16-31}

    v8f acc = {};
#pragma unroll
    for (int tap = 0; tap < 9; ++tap) {
        int r  = tap / 3;
        int dx = tap % 3;
        const _Float16* arow = &tile[r][pixT * 16 + m + dx][0];
#pragma unroll
        for (int kc = 0; kc < CIN / 32; ++kc) {
            union { v16h v; v8h h[2]; } A;
            int cb = kc * 32 + hi * 16;
            A.h[0] = *(const v8h*)(arow + cb);
            A.h[1] = *(const v8h*)(arow + cb + 8);
            union { v16h v; v8h h[2]; } Bf;
            const _Float16* bp =
                wpack + ((((size_t)tap * (CIN / 32) + kc) * NTILES + ntile) * 32 + lane) * 16;
            Bf.h[0] = *(const v8h*)(bp);
            Bf.h[1] = *(const v8h*)(bp + 8);
            acc = __builtin_amdgcn_wmma_f32_16x16x32_f16(
                false, A.v, false, Bf.v, (short)0, acc, false, false);
        }
    }

    // ---- epilogue: bias (+ReLU), store ----
    int n = ntile * 16 + (lane & 15);
    float bv = (n < COUT_REAL) ? bias[n] : 0.0f;
#pragma unroll
    for (int r2 = 0; r2 < 8; ++r2) {
        int M = pixT * 16 + r2 + hi * 8;    // pixel within row tile (C/D layout)
        int x = xbase + M;
        float v = acc[r2] + bv;
        size_t pix = ((size_t)b * HH + y) * WW + x;
        if (!FINAL) {
            v = fmaxf(v, 0.0f);
            out16[pix * COUT + n] = (_Float16)v;
        } else if (n < COUT_REAL) {
            outf[pix * COUT_REAL + n] = v;
        }
    }
}

// ---------------------------------------------------------------------------
// Classifier: GAP over f4 (4,32,32,256) -> FC 256->128 relu -> 128->128 relu
// -> 128->3 -> softmax + argmax.  One workgroup; tiny FLOPs.
// ---------------------------------------------------------------------------
__global__ __launch_bounds__(256) void classifier_kernel(const float* __restrict__ f4,
                                                         const float* __restrict__ Wc1,
                                                         const float* __restrict__ bc1,
                                                         const float* __restrict__ Wc2,
                                                         const float* __restrict__ bc2,
                                                         const float* __restrict__ Wc3,
                                                         const float* __restrict__ bc3,
                                                         float* __restrict__ pred_out,
                                                         int* __restrict__ cls_out) {
    __shared__ float gap[4][256];
    __shared__ float z1[4][128];
    __shared__ float z2[4][128];
    __shared__ float logits[4][3];
    int t = threadIdx.x;

    for (int b = 0; b < 4; ++b) {
        float s = 0.0f;
        const float* p = f4 + ((size_t)b * 1024) * 256 + t;
        for (int i = 0; i < 1024; ++i) s += p[(size_t)i * 256];
        gap[b][t] = s * (1.0f / 1024.0f);
    }
    __syncthreads();

    for (int o = t; o < 512; o += 256) {
        int b = o >> 7, j = o & 127;
        float s = bc1[j];
        for (int k = 0; k < 256; ++k) s += gap[b][k] * Wc1[k * 128 + j];
        z1[b][j] = fmaxf(s, 0.0f);
    }
    __syncthreads();

    for (int o = t; o < 512; o += 256) {
        int b = o >> 7, j = o & 127;
        float s = bc2[j];
        for (int k = 0; k < 128; ++k) s += z1[b][k] * Wc2[k * 128 + j];
        z2[b][j] = fmaxf(s, 0.0f);
    }
    __syncthreads();

    if (t < 12) {
        int b = t / 3, j = t % 3;
        float s = bc3[j];
        for (int k = 0; k < 128; ++k) s += z2[b][k] * Wc3[k * 3 + j];
        logits[b][j] = s;
    }
    __syncthreads();

    if (t < 4) {
        float l0 = logits[t][0], l1 = logits[t][1], l2 = logits[t][2];
        float mx = fmaxf(l0, fmaxf(l1, l2));
        float e0 = __expf(l0 - mx), e1 = __expf(l1 - mx), e2 = __expf(l2 - mx);
        float inv = 1.0f / (e0 + e1 + e2);
        pred_out[t * 3 + 0] = e0 * inv;
        pred_out[t * 3 + 1] = e1 * inv;
        pred_out[t * 3 + 2] = e2 * inv;
        int c = 0; float best = l0;
        if (l1 > best) { best = l1; c = 1; }
        if (l2 > best) { best = l2; c = 2; }
        cls_out[t] = c;
    }
}

// ---------------------------------------------------------------------------
// Gather: per-pixel select class channels from the full conv3 maps.
// ---------------------------------------------------------------------------
__global__ __launch_bounds__(256) void gather_kernel(const float* __restrict__ off_full,
                                                     const float* __restrict__ w_full,
                                                     const int* __restrict__ cls,
                                                     float* __restrict__ out_off,
                                                     float* __restrict__ out_w) {
    int i = blockIdx.x * blockDim.x + threadIdx.x;
    if (i >= (int)NPIX) return;
    int b = i >> 18;                 // / (512*512)
    int c = cls[b];
    size_t p = (size_t)i;
    out_off[p * 2 + 0] = off_full[p * 6 + c * 2 + 0];
    out_off[p * 2 + 1] = off_full[p * 6 + c * 2 + 1];
    out_w[p] = w_full[p * 3 + c];
}

// ---------------------------------------------------------------------------
extern "C" void kernel_launch(void* const* d_in, const int* in_sizes, int n_in,
                              void* d_out, int out_size, void* d_ws, size_t ws_size,
                              hipStream_t stream) {
    (void)in_sizes; (void)n_in; (void)out_size; (void)ws_size;

    const float* x0  = (const float*)d_in[0];
    const float* f4  = (const float*)d_in[1];
    const float* Wo1 = (const float*)d_in[2];  const float* bo1 = (const float*)d_in[3];
    const float* Wo2 = (const float*)d_in[4];  const float* bo2 = (const float*)d_in[5];
    const float* Wo3 = (const float*)d_in[6];  const float* bo3 = (const float*)d_in[7];
    const float* Ww1 = (const float*)d_in[8];  const float* bw1 = (const float*)d_in[9];
    const float* Ww2 = (const float*)d_in[10]; const float* bw2 = (const float*)d_in[11];
    const float* Ww3 = (const float*)d_in[12]; const float* bw3 = (const float*)d_in[13];
    const float* Wc1 = (const float*)d_in[14]; const float* bc1 = (const float*)d_in[15];
    const float* Wc2 = (const float*)d_in[16]; const float* bc2 = (const float*)d_in[17];
    const float* Wc3 = (const float*)d_in[18]; const float* bc3 = (const float*)d_in[19];

    char* ws = (char*)d_ws;
    _Float16* buf1  = (_Float16*)(ws + WS_BUF1);
    _Float16* buf2  = (_Float16*)(ws + WS_BUF2);
    float*    offF  = (float*)(ws + WS_OFFF);
    float*    wFull = (float*)(ws + WS_WFULL);
    _Float16* pO1 = (_Float16*)(ws + WS_PACK_O1);
    _Float16* pO2 = (_Float16*)(ws + WS_PACK_O2);
    _Float16* pO3 = (_Float16*)(ws + WS_PACK_O3);
    _Float16* pW1 = (_Float16*)(ws + WS_PACK_W1);
    _Float16* pW2 = (_Float16*)(ws + WS_PACK_W2);
    _Float16* pW3 = (_Float16*)(ws + WS_PACK_W3);
    int*      cls = (int*)(ws + WS_CLS);

    float* outOff  = (float*)d_out;                 // (4,512,512,2)
    float* outW    = outOff + (size_t)2 * NPIX;     // (4,512,512,1)
    float* outPred = outW + (size_t)NPIX;           // (4,3)

    // ---- pack weights into WMMA B-fragment order ----
    pack_w_kernel<<<dim3((9 * 32 * 64 + 255) / 256), 256, 0, stream>>>(Wo1, pO1, 32, 64, 4);
    pack_w_kernel<<<dim3((9 * 64 * 64 + 255) / 256), 256, 0, stream>>>(Wo2, pO2, 64, 64, 4);
    pack_w_kernel<<<dim3((9 * 64 * 16 + 255) / 256), 256, 0, stream>>>(Wo3, pO3, 64, 6, 1);
    pack_w_kernel<<<dim3((9 * 32 * 64 + 255) / 256), 256, 0, stream>>>(Ww1, pW1, 32, 64, 4);
    pack_w_kernel<<<dim3((9 * 64 * 64 + 255) / 256), 256, 0, stream>>>(Ww2, pW2, 64, 64, 4);
    pack_w_kernel<<<dim3((9 * 64 * 16 + 255) / 256), 256, 0, stream>>>(Ww3, pW3, 64, 3, 1);

    dim3 g1(BATCH * HH * (WW / 32));    // 32768 workgroups (32-px x 64-ch tiles)
    dim3 g3(BATCH * HH * (WW / 128));   // 8192 workgroups (128-px x 16-ch tiles)

    // offsets branch
    conv3x3_kernel<32, 2, 4, 64, false, true ><<<g1, 256, 0, stream>>>(x0,   pO1, bo1, buf1, nullptr);
    conv3x3_kernel<64, 2, 4, 64, false, false><<<g1, 256, 0, stream>>>(buf1, pO2, bo2, buf2, nullptr);
    conv3x3_kernel<64, 8, 1,  6, true,  false><<<g3, 256, 0, stream>>>(buf2, pO3, bo3, nullptr, offF);
    // weights branch
    conv3x3_kernel<32, 2, 4, 64, false, true ><<<g1, 256, 0, stream>>>(x0,   pW1, bw1, buf1, nullptr);
    conv3x3_kernel<64, 2, 4, 64, false, false><<<g1, 256, 0, stream>>>(buf1, pW2, bw2, buf2, nullptr);
    conv3x3_kernel<64, 8, 1,  3, true,  false><<<g3, 256, 0, stream>>>(buf2, pW3, bw3, nullptr, wFull);

    // classifier + gather
    classifier_kernel<<<1, 256, 0, stream>>>(f4, Wc1, bc1, Wc2, bc2, Wc3, bc3, outPred, cls);
    gather_kernel<<<dim3(((int)NPIX + 255) / 256), 256, 0, stream>>>(offF, wFull, cls, outOff, outW);
}
